// HiddenLayer_86148454023480
// MI455X (gfx1250) — matrix operand order repaired and checked by reading, not compiled, not run on hardware
//
#include <hip/hip_runtime.h>

// ---------------------------------------------------------------------------
// MI455X (gfx1250) fused MLP towers:
//   h   = relu(x @ W1 + b1)        [T,B,H]
//   out = cumsum_t(h @ W2 + b2)    [T,B,O]
// Pipeline:
//   1) one-shot fp32->bf16 convert of x; fp32->bf16 + transpose of W1,W2
//   2) bf16 WMMA GEMMs (v_wmma_f32_16x16x32_bf16, fp32 accum). Staging uses
//      gfx1250 async global->LDS copies (ASYNCcnt) with LDS double-buffering
//      when available; falls back to register-staged global->LDS otherwise.
//   3) bandwidth-bound cumsum over the 8 towers
// T=8, B=4096, I=1024, H=2048, O=512.
// ---------------------------------------------------------------------------

typedef __attribute__((ext_vector_type(16))) __bf16      v16bf;
typedef __attribute__((ext_vector_type(8)))  float       v8f;
typedef __attribute__((ext_vector_type(4)))  unsigned int u32x4;
typedef int gv4i __attribute__((vector_size(16)));   // matches builtin pointee type

#define BM 128
#define BN 128
#define BK 64
#define LDT 72   // LDS row stride (ushort): 64 + 8 pad, 144B rows (16B aligned)

#if defined(__has_builtin)
#if __has_builtin(__builtin_amdgcn_global_load_async_to_lds_b128)
#define USE_ASYNC_COPY 1
#endif
#endif
#ifndef USE_ASYNC_COPY
#define USE_ASYNC_COPY 0
#endif

#define AS1 __attribute__((address_space(1)))
#define AS3 __attribute__((address_space(3)))

#if USE_ASYNC_COPY
// One 16B async global->LDS copy per lane (GLOBAL_LOAD_ASYNC_TO_LDS_B128).
__device__ __forceinline__ void async_ld16(unsigned short* lds, const unsigned short* g) {
    __builtin_amdgcn_global_load_async_to_lds_b128(
        (AS1 gv4i*)(unsigned short*)g, (AS3 gv4i*)lds, 0, 0);
}
__device__ __forceinline__ void wait_async0() {
#if __has_builtin(__builtin_amdgcn_s_wait_asynccnt)
    __builtin_amdgcn_s_wait_asynccnt(0);
#else
    asm volatile("s_wait_asynccnt 0x0" ::: "memory");
#endif
}
#endif

__device__ __forceinline__ unsigned short f32_to_bf16_bits(float f) {
    union { float f; unsigned int u; } v; v.f = f;
    unsigned int r = v.u + 0x7FFFu + ((v.u >> 16) & 1u);   // RNE
    return (unsigned short)(r >> 16);
}

// Pack bf16(a) into [15:0], bf16(b) into [31:16] with one v_perm_b32.
__device__ __forceinline__ unsigned int pack2_bf16(float a, float b) {
    union { float f; unsigned int u; } x, y; x.f = a; y.f = b;
    unsigned int ra = x.u + 0x7FFFu + ((x.u >> 16) & 1u);
    unsigned int rb = y.u + 0x7FFFu + ((y.u >> 16) & 1u);
    return __builtin_amdgcn_perm(rb, ra, 0x07060302u);
}

// ---------------------------------------------------------------------------
// Elementwise fp32 -> bf16 (x). Grid covers n exactly, 8 elems/thread.
// ---------------------------------------------------------------------------
__global__ void __launch_bounds__(256)
cvt_f32_bf16_kernel(const float* __restrict__ in, unsigned short* __restrict__ out) {
    size_t i0 = ((size_t)blockIdx.x * 256 + threadIdx.x) * 8;
    float4 a = *(const float4*)(in + i0);
    float4 b = *(const float4*)(in + i0 + 4);
    u32x4 p;
    p[0] = pack2_bf16(a.x, a.y);
    p[1] = pack2_bf16(a.z, a.w);
    p[2] = pack2_bf16(b.x, b.y);
    p[3] = pack2_bf16(b.z, b.w);
    *(u32x4*)(out + i0) = p;
}

// ---------------------------------------------------------------------------
// fp32 [T,K,N] -> bf16 [T,N,K] (transpose via LDS 64x64 tile).
// ---------------------------------------------------------------------------
#define TT  64
#define TLD 66   // odd bank stride (33 banks) for conflict-light column reads

__global__ void __launch_bounds__(256)
cvt_transpose_bf16_kernel(const float* __restrict__ W, unsigned short* __restrict__ WT,
                          int K, int N) {
    __shared__ __align__(16) unsigned short tile[TT * TLD];
    const int t  = blockIdx.z;
    const int n0 = blockIdx.x * TT;
    const int k0 = blockIdx.y * TT;
    const float* Wt = W + (size_t)t * K * N;
    unsigned short* WTt = WT + (size_t)t * N * K;
    const int tid = threadIdx.x;

    #pragma unroll
    for (int i = 0; i < 4; i++) {                 // 1024 float4s
        int e  = tid + i * 256;
        int r  = e >> 4;                          // k row 0..63
        int c4 = (e & 15) << 2;                   // n col
        float4 v = *(const float4*)(Wt + (size_t)(k0 + r) * N + n0 + c4);
        unsigned int* p = (unsigned int*)(tile + r * TLD + c4);
        p[0] = pack2_bf16(v.x, v.y);
        p[1] = pack2_bf16(v.z, v.w);
    }
    __syncthreads();
    #pragma unroll
    for (int i = 0; i < 8; i++) {                 // 2048 uint stores
        int e  = tid + i * 256;
        int nr = e >> 5;                          // n row 0..63
        int c2 = (e & 31) << 1;                   // k pair
        unsigned int lo = tile[(c2 + 0) * TLD + nr];
        unsigned int hi = tile[(c2 + 1) * TLD + nr];
        *(unsigned int*)(WTt + (size_t)(n0 + nr) * K + k0 + c2) = lo | (hi << 16);
    }
}

// ---------------------------------------------------------------------------
// bf16 GEMM: C[M,N] = A[M,K] . BT[N,K]^T + bias, per tower (blockIdx.z).
// Block tile 128x128, BK=64; 8 waves as 2(m) x 4(n); wave tile 64x32 = 4x2
// WMMA 16x16 tiles. Async path: double-buffered LDS, tile k+1 streamed by the
// async copy engine (ASYNCcnt) while tile k feeds the WMMA pipe.
// ---------------------------------------------------------------------------
template <bool RELU_BF16_OUT>
__global__ void __launch_bounds__(256)
gemm_bf16_kernel(const unsigned short* __restrict__ Ag,   // [T,M,K] bf16
                 const unsigned short* __restrict__ BTg,  // [T,N,K] bf16
                 const float* __restrict__ biasg,         // [T,N]
                 void* __restrict__ Cg,                   // [T,M,N]
                 int M, int K, int N) {
#if USE_ASYNC_COPY
    __shared__ __align__(16) unsigned short sA[2][BM * LDT];
    __shared__ __align__(16) unsigned short sB[2][BN * LDT];
#else
    __shared__ __align__(16) unsigned short sA[1][BM * LDT];
    __shared__ __align__(16) unsigned short sB[1][BN * LDT];
#endif

    const int t  = blockIdx.z;
    const int m0 = blockIdx.y * BM;
    const int n0 = blockIdx.x * BN;
    const unsigned short* At  = Ag  + (size_t)t * M * K;
    const unsigned short* BTt = BTg + (size_t)t * N * K;
    const float* bt = biasg + (size_t)t * N;

    const int tid   = threadIdx.x;
    const int wave  = tid >> 5;
    const int lane  = tid & 31;
    const int wm    = (wave & 1) << 6;   // 0,64
    const int wn    = (wave >> 1) << 5;  // 0,32,64,96
    const int lr    = lane & 15;
    const int khalf = lane >> 4;

    // Staging: 128x64 bf16 tile = 1024 x 16B; 4 per thread, fixed column c8.
    const int c8 = (tid & 7) << 3;       // ushort offset within K-slab
    const int r0 = tid >> 3;             // row 0..31, +32 per iter
    const unsigned short* Abase = At  + (size_t)m0 * K + c8;
    const unsigned short* Bbase = BTt + (size_t)n0 * K + c8;

    v8f acc[4][2];
    #pragma unroll
    for (int im = 0; im < 4; im++)
        #pragma unroll
        for (int jn = 0; jn < 2; jn++)
            #pragma unroll
            for (int e = 0; e < 8; e++) acc[im][jn][e] = 0.0f;

    // One BK=64 compute stage from LDS buffers cA/cB: 2 K-steps x 8 WMMA.
    auto compute_stage = [&](const unsigned short* cA, const unsigned short* cB) {
        #pragma unroll
        for (int s = 0; s < 2; s++) {
            v16bf af[4], bfr[2];
            #pragma unroll
            for (int im = 0; im < 4; im++) {
                // A 16x32 frag: lane group g holds K = {g*8..+7, 16+g*8..+7}
                const unsigned short* p = cA + (wm + im * 16 + lr) * LDT + s * 32;
                union { u32x4 u[2]; v16bf b; } f;
                f.u[0] = *(const u32x4*)(p + khalf * 8);
                f.u[1] = *(const u32x4*)(p + 16 + khalf * 8);
                af[im] = f.b;
            }
            #pragma unroll
            for (int jn = 0; jn < 2; jn++) {
                // B 32x16 frag (staged [n][k]): lane group g holds K = {g*8..+7, 16+g*8..+7}
                const unsigned short* p = cB + (wn + jn * 16 + lr) * LDT + s * 32;
                union { u32x4 u[2]; v16bf b; } f;
                f.u[0] = *(const u32x4*)(p + khalf * 8);
                f.u[1] = *(const u32x4*)(p + 16 + khalf * 8);
                bfr[jn] = f.b;
            }
            #pragma unroll
            for (int im = 0; im < 4; im++)
                #pragma unroll
                for (int jn = 0; jn < 2; jn++)
                    acc[im][jn] = __builtin_amdgcn_wmma_f32_16x16x32_bf16(
                        false, af[im], false, bfr[jn], (short)0, acc[im][jn], false, false);
        }
    };

#if USE_ASYNC_COPY
    // Double-buffered async pipeline:
    //   wait(asynccnt==0); barrier; issue tile k+1 -> other buffer; compute k;
    //   barrier (all waves done reading buf before it is re-targeted).
    auto issue = [&](int k0, int buf) {
        #pragma unroll
        for (int i = 0; i < 4; i++) {
            int row = r0 + 32 * i;
            async_ld16(&sA[buf][row * LDT + c8], Abase + (size_t)row * K + k0);
            async_ld16(&sB[buf][row * LDT + c8], Bbase + (size_t)row * K + k0);
        }
    };
    const int nIter = K / BK;
    issue(0, 0);
    for (int it = 0; it < nIter; ++it) {
        const int cur = it & 1;
        wait_async0();
        __syncthreads();
        if (it + 1 < nIter) issue((it + 1) * BK, cur ^ 1);
        compute_stage(sA[cur], sB[cur]);
        __syncthreads();
    }
#else
    // Fallback: register-staged pipeline (fetch k+1 while computing k).
    u32x4 ra[4], rb[4];
    auto fetch = [&](int k0) {
        #pragma unroll
        for (int i = 0; i < 4; i++) {
            int row = r0 + 32 * i;
            ra[i] = *(const u32x4*)(Abase + (size_t)row * K + k0);
            rb[i] = *(const u32x4*)(Bbase + (size_t)row * K + k0);
        }
    };
    fetch(0);
    for (int k0 = 0; k0 < K; k0 += BK) {
        #pragma unroll
        for (int i = 0; i < 4; i++) {
            int row = r0 + 32 * i;
            *(u32x4*)(sA[0] + row * LDT + c8) = ra[i];
            *(u32x4*)(sB[0] + row * LDT + c8) = rb[i];
        }
        __syncthreads();
        if (k0 + BK < K) fetch(k0 + BK);
        compute_stage(sA[0], sB[0]);
        __syncthreads();
    }
#endif

    // Epilogue. C layout: VGPR r, lane l -> M = (l/16)*8 + r, N = l%16.
    #pragma unroll
    for (int im = 0; im < 4; im++) {
        int mbase = m0 + wm + im * 16 + khalf * 8;
        #pragma unroll
        for (int jn = 0; jn < 2; jn++) {
            int n = n0 + wn + jn * 16 + lr;
            float bias = bt[n];
            if constexpr (RELU_BF16_OUT) {
                unsigned short* C = (unsigned short*)Cg + (size_t)t * M * N;
                #pragma unroll
                for (int r = 0; r < 8; r++) {
                    float v = acc[im][jn][r] + bias;
                    v = v > 0.0f ? v : 0.0f;
                    C[(size_t)(mbase + r) * N + n] = f32_to_bf16_bits(v);
                }
            } else {
                float* C = (float*)Cg + (size_t)t * M * N;
                #pragma unroll
                for (int r = 0; r < 8; r++)
                    C[(size_t)(mbase + r) * N + n] = acc[im][jn][r] + bias;
            }
        }
    }
}

// ---------------------------------------------------------------------------
// Cumulative sum across the T(=8) tower axis; pure bandwidth, float4 lanes.
// ---------------------------------------------------------------------------
__global__ void __launch_bounds__(256)
cumsum_t_kernel(const float* __restrict__ tmp, float* __restrict__ out, int n4, int T) {
    int idx = blockIdx.x * 256 + threadIdx.x;
    if (idx >= n4) return;
    float4 s = {0.0f, 0.0f, 0.0f, 0.0f};
    for (int t = 0; t < T; t++) {
        float4 v = *(const float4*)(tmp + ((size_t)t * n4 + idx) * 4);
        s.x += v.x; s.y += v.y; s.z += v.z; s.w += v.w;
        *(float4*)(out + ((size_t)t * n4 + idx) * 4) = s;
    }
}

extern "C" void kernel_launch(void* const* d_in, const int* in_sizes, int n_in,
                              void* d_out, int out_size, void* d_ws, size_t ws_size,
                              hipStream_t stream) {
    const float* x  = (const float*)d_in[0];   // [T,B,I]
    const float* W1 = (const float*)d_in[1];   // [T,I,H]
    const float* b1 = (const float*)d_in[2];   // [T,H]
    const float* W2 = (const float*)d_in[3];   // [T,H,O]
    const float* b2 = (const float*)d_in[4];   // [T,O]
    float* out = (float*)d_out;                // [T,B,O]

    const int T = 8, Bm = 4096, I = 1024, H = 2048, O = 512;

    // Workspace layout (252 MB):
    //   [0)       h bf16    [T,B,H]  134.2 MB
    //   [off_x)   x bf16    [T,B,I]   67.1 MB  -- aliased later by fp32 tower
    //                                            outputs (identical size)
    //   [off_w1)  W1^T bf16 [T,H,I]   33.6 MB
    //   [off_w2)  W2^T bf16 [T,O,H]   16.8 MB
    char* ws = (char*)d_ws;
    unsigned short* hbuf = (unsigned short*)ws;
    size_t off_x  = (size_t)T * Bm * H * 2;
    unsigned short* xbf = (unsigned short*)(ws + off_x);
    float* tmp = (float*)(ws + off_x);                       // alias (gemm1 done first)
    size_t off_w1 = off_x + (size_t)T * Bm * I * 2;
    unsigned short* w1t = (unsigned short*)(ws + off_w1);
    size_t off_w2 = off_w1 + (size_t)T * I * H * 2;
    unsigned short* w2t = (unsigned short*)(ws + off_w2);

    dim3 blk(256);

    // 1) precision conversion + weight transpose (one-shot, bandwidth bound)
    cvt_f32_bf16_kernel<<<((size_t)T * Bm * I) / 2048, blk, 0, stream>>>(x, xbf);
    cvt_transpose_bf16_kernel<<<dim3(H / TT, I / TT, T), blk, 0, stream>>>(W1, w1t, I, H);
    cvt_transpose_bf16_kernel<<<dim3(O / TT, H / TT, T), blk, 0, stream>>>(W2, w2t, H, O);

    // 2) WMMA GEMMs
    gemm_bf16_kernel<true ><<<dim3(H / BN, Bm / BM, T), blk, 0, stream>>>(xbf, w1t, b1, hbuf, Bm, I, H);
    gemm_bf16_kernel<false><<<dim3(O / BN, Bm / BM, T), blk, 0, stream>>>(hbuf, w2t, b2, tmp, Bm, H, O);

    // 3) tower prefix sum
    int n4 = (Bm * O) / 4;
    cumsum_t_kernel<<<(n4 + 255) / 256, blk, 0, stream>>>(tmp, out, n4, T);
}
